// qkvblock_18743237280781
// MI455X (gfx1250) — matrix operand order repaired and checked by reading, not compiled, not run on hardware
//
#include <hip/hip_runtime.h>
#include <hip/hip_bf16.h>

// ---------------------------------------------------------------------------
// MI455X (gfx1250): wave32, v_wmma_f32_16x16x32_bf16 for all channel GEMMs.
// - 16x64 output tile per wave (4 accumulators, shared A fragment)
// - ping-pong double-buffered K loops (no loop-carried copies, partial waits)
// - qq / k-half stored channel-major via LDS transpose so the Gram GEMM
//   (contraction over 65536 pixels) uses contiguous b128 fragment loads
// - attn+po fused into one GEMM via Mfull = po_w @ attn
// - branch-1/2 tails computed only at the ::2 / ::4 sampled pixels
// NOTE: all GEMM K values are multiples of 64 (192/128/64, gram chunk 2048).
// ---------------------------------------------------------------------------

typedef __bf16 bf16;
typedef __attribute__((ext_vector_type(16))) __bf16 v16bf;
typedef __attribute__((ext_vector_type(8)))  float  v8f;

#define Bsz 2
#define Hf  256
#define HWf 65536
#define Cch 64
#define C3  192
#define C6  384
#define NCHG 32          // gram split-K chunks
#define GCHUNK (HWf / NCHG)

// ---- WMMA fragment loaders (ISA 7.12.2 16-bit layouts, wave32) -------------

// A (16x32 bf16): lane=(m,ksel). element s: K=ksel*16+s (contiguous)
__device__ __forceinline__ v16bf loadA_row(const bf16* __restrict__ rowptr, int ksel) {
  v16bf f;
  const bf16* p  = rowptr + ksel * 16;
  const bf16* p2 = rowptr + ksel * 16 + 8;
#pragma unroll
  for (int j = 0; j < 4; ++j) { f[2*j] = p[2*j];  f[2*j+1] = p[2*j+1]; }
#pragma unroll
  for (int j = 0; j < 4; ++j) { f[8+2*j] = p2[2*j]; f[8+2*j+1] = p2[2*j+1]; }
  return f;
}

// B (32x16 bf16) row-major source W[n][k]: lane=(n,ksel); 16 contiguous bf16.
__device__ __forceinline__ v16bf loadB_row(const bf16* __restrict__ p) {
  v16bf f;
#pragma unroll
  for (int j = 0; j < 16; ++j) f[j] = p[j];
  return f;
}

__device__ __forceinline__ v8f wmma_bf16(v16bf a, v16bf b, v8f c) {
  return __builtin_amdgcn_wmma_f32_16x16x32_bf16(false, a, false, b, (short)0, c,
                                                 false, false);
}

// ---- weight conversion -----------------------------------------------------
__global__ void f2bf_kernel(const float* __restrict__ s, bf16* __restrict__ d, int n) {
  int i = blockIdx.x * blockDim.x + threadIdx.x;
  if (i < n) d[i] = (bf16)s[i];
}

// ---- LayerNorm over 64 channels --------------------------------------------
__device__ __forceinline__ void ln64(const float* __restrict__ src, size_t cstride,
                                     const float* __restrict__ w,
                                     const float* __restrict__ b,
                                     bf16* __restrict__ dst) {
  float v[64]; float s = 0.f;
#pragma unroll
  for (int c = 0; c < 64; ++c) { v[c] = src[(size_t)c * cstride]; s += v[c]; }
  float mu = s * (1.f / 64.f), s2 = 0.f;
#pragma unroll
  for (int c = 0; c < 64; ++c) { float d = v[c] - mu; s2 += d * d; }
  float inv = rsqrtf(s2 * (1.f / 64.f) + 1e-6f);
#pragma unroll
  for (int c = 0; c < 64; ++c) dst[c] = (bf16)(w[c] * ((v[c] - mu) * inv) + b[c]);
}

// LN of enc0 / up2(enc1) / up4(enc2) -> concat [p][192] bf16 (NHWC)
__global__ void ln_concat_kernel(const float* __restrict__ e0, const float* __restrict__ e1,
                                 const float* __restrict__ e2,
                                 const float* __restrict__ w0, const float* __restrict__ b0,
                                 const float* __restrict__ w1, const float* __restrict__ b1,
                                 const float* __restrict__ w2, const float* __restrict__ b2,
                                 bf16* __restrict__ xcat) {
  int p = blockIdx.x * blockDim.x + threadIdx.x;
  if (p >= Bsz * HWf) return;
  int b = p >> 16, hw = p & (HWf - 1);
  int hh = hw >> 8, ww = hw & 255;
  bf16* out = xcat + (size_t)p * C3;
  ln64(e0 + (size_t)b * 64 * HWf + hw, HWf, w0, b0, out);
  ln64(e1 + (size_t)b * 64 * 16384 + (size_t)(hh >> 1) * 128 + (ww >> 1), 16384, w1, b1, out + 64);
  ln64(e2 + (size_t)b * 64 * 4096  + (size_t)(hh >> 2) * 64  + (ww >> 2), 4096,  w2, b2, out + 128);
}

// LN over contiguous f32 rows [M][64] -> bf16 rows
__global__ void ln_rows_kernel(const float* __restrict__ Y, const float* __restrict__ w,
                               const float* __restrict__ b, bf16* __restrict__ o, int M) {
  int r = blockIdx.x * blockDim.x + threadIdx.x;
  if (r < M) ln64(Y + (size_t)r * 64, 1, w, b, o + (size_t)r * 64);
}

// ---- generic GEMM: 16 rows x 64 cols per wave, ping-pong K (K%64==0) --------
__global__ __launch_bounds__(256)
void gemm_bias_bf16(const bf16* __restrict__ A, int lda, int aoff,
                    const bf16* __restrict__ W, int ldw,
                    const float* __restrict__ bias,
                    bf16* __restrict__ O, int ldo, int M, int N, int K) {
  int lane = threadIdx.x & 31;
  int wave = blockIdx.x * (blockDim.x >> 5) + (threadIdx.x >> 5);
  int ngroups = N >> 6;                           // 64-col groups
  int mtile = wave / ngroups, g = wave % ngroups;
  int row0 = mtile << 4;
  if (row0 >= M) return;
  int m = lane & 15, ksel = lane >> 4;
  const bf16* arow = A + (size_t)(row0 + m) * lda + aoff;
  const bf16* wrow[4];
#pragma unroll
  for (int t = 0; t < 4; ++t)
    wrow[t] = W + (size_t)((g << 6) + (t << 4) + m) * ldw + ksel * 16;
  __builtin_prefetch(arow + K, 0, 0);             // global_prefetch_b8
  v8f acc[4] = {};
  // ping-pong prologue: two K-steps in flight
  v16bf a0 = loadA_row(arow, ksel), a1 = loadA_row(arow + 32, ksel);
  v16bf b0[4], b1[4];
#pragma unroll
  for (int t = 0; t < 4; ++t) { b0[t] = loadB_row(wrow[t]); b1[t] = loadB_row(wrow[t] + 32); }
  for (int kb = 64; kb < K; kb += 64) {
#pragma unroll
    for (int t = 0; t < 4; ++t) acc[t] = wmma_bf16(a0, b0[t], acc[t]);
    a0 = loadA_row(arow + kb, ksel);
#pragma unroll
    for (int t = 0; t < 4; ++t) b0[t] = loadB_row(wrow[t] + kb);
#pragma unroll
    for (int t = 0; t < 4; ++t) acc[t] = wmma_bf16(a1, b1[t], acc[t]);
    a1 = loadA_row(arow + kb + 32, ksel);
#pragma unroll
    for (int t = 0; t < 4; ++t) b1[t] = loadB_row(wrow[t] + kb + 32);
  }
#pragma unroll
  for (int t = 0; t < 4; ++t) acc[t] = wmma_bf16(a0, b0[t], acc[t]);
#pragma unroll
  for (int t = 0; t < 4; ++t) acc[t] = wmma_bf16(a1, b1[t], acc[t]);
  // epilogue
#pragma unroll
  for (int t = 0; t < 4; ++t) {
    int o = (g << 6) + (t << 4) + m;              // lane&15 is the output col
    float bv = bias ? bias[o] : 0.f;
#pragma unroll
    for (int r = 0; r < 8; ++r) {
      int row = row0 + r + (ksel << 3);
      O[(size_t)row * ldo + o] = (bf16)(acc[t][r] + bv);
    }
  }
}

// ---- depthwise 3x3, SAME, NHWC bf16 ----------------------------------------
__global__ void dwconv3_kernel(const bf16* __restrict__ X, const float* __restrict__ W,
                               const float* __restrict__ bias, bf16* __restrict__ O,
                               int Cc, long total) {
  long idx = (long)blockIdx.x * blockDim.x + threadIdx.x;
  if (idx >= total) return;
  int c = (int)(idx % Cc); long t = idx / Cc;
  int ww = (int)(t % Hf); t /= Hf;
  int hh = (int)(t % Hf); int b = (int)(t / Hf);
  float acc = bias[c];
  const float* wc = W + c * 9;
#pragma unroll
  for (int dy = 0; dy < 3; ++dy) {
    int y = hh + dy - 1; if ((unsigned)y >= Hf) continue;
#pragma unroll
    for (int dx = 0; dx < 3; ++dx) {
      int x = ww + dx - 1; if ((unsigned)x >= Hf) continue;
      acc += (float)X[(((size_t)b * Hf + y) * Hf + x) * Cc + c] * wc[dy * 3 + dx];
    }
  }
  O[idx] = (bf16)acc;
}

// ---- LDS-tiled bf16 transpose: [b*HW+p][ldi] cols[coff..coff+Cn) -> [b][c][p]
__global__ void transpose_bf16(const bf16* __restrict__ in, int ldi, int coff,
                               bf16* __restrict__ out, int Cn) {
  __shared__ bf16 tile[32][33];
  int b = blockIdx.z;
  int p0 = blockIdx.x << 5, c0 = blockIdx.y << 5;
  int tx = threadIdx.x, ty = threadIdx.y;           // 32 x 8
  for (int i = ty; i < 32; i += 8)
    tile[i][tx] = in[((size_t)b * HWf + p0 + i) * ldi + coff + c0 + tx];
  __syncthreads();
  for (int i = ty; i < 32; i += 8)
    out[((size_t)(b * Cn + c0 + i)) * HWf + p0 + tx] = tile[tx][i];
}

// ---- per-channel spatial sum-of-squares on channel-major bf16 ---------------
__global__ void sqnorm_partial_cm(const bf16* __restrict__ X, float* __restrict__ part,
                                  int chunkPix) {
  int c = threadIdx.x, Cn = blockDim.x;
  int ch = blockIdx.x, b = blockIdx.y;
  const bf16* p = X + ((size_t)(b * Cn + c)) * HWf + (size_t)ch * chunkPix;
  float s = 0.f;
  for (int i = 0; i < chunkPix; ++i) { float v = (float)p[i]; s += v * v; }
  part[((size_t)(b * gridDim.x + ch)) * Cn + c] = s;
}

__global__ void sqnorm_finalize(const float* __restrict__ part, float* __restrict__ nrm,
                                int Cn, int nch) {
  int i = blockIdx.x * blockDim.x + threadIdx.x;
  if (i >= Bsz * Cn) return;
  int b = i / Cn, c = i % Cn;
  float s = 0.f;
  for (int ch = 0; ch < nch; ++ch) s += part[((size_t)(b * nch + ch)) * Cn + c];
  nrm[i] = sqrtf(s);
}

// ---- Gram on channel-major operands: contiguous b128 loads, ping-pong K -----
// G[b,br,h][32][96] = sum_p qq_cm[c][p] * kk_cm[d][p], split-K over chunks.
__global__ __launch_bounds__(384)
void gram_kernel(const bf16* __restrict__ QQ0, const bf16* __restrict__ QQ1,
                 const bf16* __restrict__ QQ2, const bf16* __restrict__ KK,
                 float* __restrict__ gpart) {
  int lane = threadIdx.x & 31, wave = threadIdx.x >> 5;   // 12 waves
  int mtile = wave / 6, ntile = wave % 6;
  int ch = blockIdx.x, b = blockIdx.y, z = blockIdx.z;
  int br = z >> 1, h = z & 1;
  const bf16* QQ = (br == 0) ? QQ0 : (br == 1) ? QQ1 : QQ2;
  int m = lane & 15, ksel = lane >> 4;
  int cq = h * 32 + mtile * 16, ck = h * 96 + ntile * 16;
  size_t pbase = (size_t)ch * GCHUNK;
  const bf16* arow = QQ + ((size_t)(b * Cch + cq + m)) * HWf + pbase;
  const bf16* brow = KK + ((size_t)(b * C3 + ck + m)) * HWf + pbase + ksel * 16;
  v8f acc = {};
  v16bf a0 = loadA_row(arow, ksel),      bb0 = loadB_row(brow);
  v16bf a1 = loadA_row(arow + 32, ksel), bb1 = loadB_row(brow + 32);
  for (int kb = 64; kb < GCHUNK; kb += 64) {
    acc = wmma_bf16(a0, bb0, acc);
    a0 = loadA_row(arow + kb, ksel); bb0 = loadB_row(brow + kb);
    acc = wmma_bf16(a1, bb1, acc);
    a1 = loadA_row(arow + kb + 32, ksel); bb1 = loadB_row(brow + kb + 32);
  }
  acc = wmma_bf16(a0, bb0, acc);
  acc = wmma_bf16(a1, bb1, acc);
  size_t tb = ((size_t)(((b * 3 + br) * 2 + h) * gridDim.x + ch)) * (32 * 96);
#pragma unroll
  for (int r = 0; r < 8; ++r) {
    int mr = mtile * 16 + r + (ksel << 3);
    gpart[tb + (size_t)mr * 96 + ntile * 16 + m] = acc[r];
  }
}

// ---- reduce partials, scale by temp/(|q||k|), relu, softmax, Mfull = po@attn
__global__ void attn_mfull_kernel(const float* __restrict__ gpart, int nch,
                                  const float* __restrict__ qn0, const float* __restrict__ qn1,
                                  const float* __restrict__ qn2, const float* __restrict__ kn,
                                  const float* __restrict__ t0, const float* __restrict__ t1,
                                  const float* __restrict__ t2,
                                  const float* __restrict__ po0, const float* __restrict__ po1,
                                  bf16* __restrict__ MF) {
  int z = blockIdx.x;                 // b*6 + br*2 + h
  int h = z & 1, br = (z >> 1) % 3, b = z / 6;
  int d = threadIdx.x;                // 0..95
  __shared__ float attn[32][96];
  const float* qn = (br == 0) ? qn0 : (br == 1) ? qn1 : qn2;
  const float* tp = (br == 0) ? t0  : (br == 1) ? t1  : t2;
  float nk = fmaxf(kn[b * C3 + h * 96 + d], 1e-12f);
  float ts = tp[h];
  size_t gb = ((size_t)((b * 3 + br) * 2 + h)) * nch * (32 * 96);
  for (int c = 0; c < 32; ++c) {
    float s = 0.f;
    for (int ch = 0; ch < nch; ++ch) s += gpart[gb + (size_t)ch * (32 * 96) + c * 96 + d];
    float nq = fmaxf(qn[b * 64 + h * 32 + c], 1e-12f);
    float a = s * ts / (nq * nk);
    attn[c][d] = a > 0.f ? a : 0.f;
  }
  __syncthreads();
  if (d < 32) {
    int c = d; float mx = attn[c][0];
    for (int j = 1; j < 96; ++j) mx = fmaxf(mx, attn[c][j]);
    float sm = 0.f;
    for (int j = 0; j < 96; ++j) { float e = __expf(attn[c][j] - mx); attn[c][j] = e; sm += e; }
    float inv = 1.f / sm;
    for (int j = 0; j < 96; ++j) attn[c][j] *= inv;
  }
  __syncthreads();
  const float* po = (br == 0) ? po0 : po1;
  for (int o = 0; o < 64; ++o) {
    float s = 0.f;
    for (int c = 0; c < 32; ++c) s += po[o * 64 + h * 32 + c] * attn[c][d];
    MF[((size_t)((b * 3 + br) * 64 + o)) * C3 + h * 96 + d] = (bf16)s;
  }
}

// ---- x = Mfull @ v + po_b ; y = enc + x*beta. 16x64 tile/wave, ping-pong ----
__global__ __launch_bounds__(256)
void attn_out_gemm(const bf16* __restrict__ KKVV, const bf16* __restrict__ MF,
                   const float* __restrict__ pob, const float* __restrict__ enc,
                   const float* __restrict__ beta, float* __restrict__ Y,
                   int br, int s, int Hs) {
  int b = blockIdx.y;
  int lane = threadIdx.x & 31;
  int wave = blockIdx.x * (blockDim.x >> 5) + (threadIdx.x >> 5);
  int HsHs = Hs * Hs;
  int row0 = wave << 4;
  if (row0 >= HsHs) return;
  int m = lane & 15, ksel = lane >> 4;
  int sp = row0 + m, hh = sp / Hs, ww = sp % Hs;
  size_t fullp = (size_t)b * HWf + (size_t)(hh * s) * Hf + (size_t)(ww * s);
  const bf16* arow = KKVV + fullp * C6 + C3;                       // v half
  const bf16* wrow[4];
#pragma unroll
  for (int t = 0; t < 4; ++t)
    wrow[t] = MF + ((size_t)((b * 3 + br) * 64 + (t << 4) + m)) * C3 + ksel * 16;
  v8f acc[4] = {};
  v16bf a0 = loadA_row(arow, ksel), a1 = loadA_row(arow + 32, ksel);
  v16bf b0[4], b1[4];
#pragma unroll
  for (int t = 0; t < 4; ++t) { b0[t] = loadB_row(wrow[t]); b1[t] = loadB_row(wrow[t] + 32); }
#pragma unroll
  for (int kb = 64; kb < C3; kb += 64) {
#pragma unroll
    for (int t = 0; t < 4; ++t) acc[t] = wmma_bf16(a0, b0[t], acc[t]);
    a0 = loadA_row(arow + kb, ksel);
#pragma unroll
    for (int t = 0; t < 4; ++t) b0[t] = loadB_row(wrow[t] + kb);
#pragma unroll
    for (int t = 0; t < 4; ++t) acc[t] = wmma_bf16(a1, b1[t], acc[t]);
    a1 = loadA_row(arow + kb + 32, ksel);
#pragma unroll
    for (int t = 0; t < 4; ++t) b1[t] = loadB_row(wrow[t] + kb + 32);
  }
#pragma unroll
  for (int t = 0; t < 4; ++t) acc[t] = wmma_bf16(a0, b0[t], acc[t]);
#pragma unroll
  for (int t = 0; t < 4; ++t) acc[t] = wmma_bf16(a1, b1[t], acc[t]);
#pragma unroll
  for (int t = 0; t < 4; ++t) {
    int o = (t << 4) + m;
    float pb = pob[o], bt = beta[o];
#pragma unroll
    for (int r = 0; r < 8; ++r) {
      int spr = row0 + r + (ksel << 3);
      float e = enc[((size_t)(b * 64 + o)) * HsHs + spr];
      Y[((size_t)(b * HsHs + spr)) * 64 + o] = e + (acc[t][r] + pb) * bt;
    }
  }
}

// ---- conv5 + residual -> NCHW output. 16x64 tile/wave -----------------------
__global__ __launch_bounds__(256)
void conv5_out_gemm(const bf16* __restrict__ G, const bf16* __restrict__ W5,
                    const float* __restrict__ b5, const float* __restrict__ Y,
                    const float* __restrict__ gamma, float* __restrict__ out,
                    int HsHs, int M) {
  int lane = threadIdx.x & 31;
  int wave = blockIdx.x * (blockDim.x >> 5) + (threadIdx.x >> 5);
  int row0 = wave << 4;
  if (row0 >= M) return;
  int m = lane & 15, ksel = lane >> 4;
  const bf16* arow = G + (size_t)(row0 + m) * 64;
  const bf16* wrow[4];
#pragma unroll
  for (int t = 0; t < 4; ++t)
    wrow[t] = W5 + (size_t)((t << 4) + m) * 64 + ksel * 16;
  v8f acc[4] = {};
  v16bf a0 = loadA_row(arow, ksel);
  v16bf a1 = loadA_row(arow + 32, ksel);
  v16bf b0[4], b1[4];
#pragma unroll
  for (int t = 0; t < 4; ++t) { b0[t] = loadB_row(wrow[t]); b1[t] = loadB_row(wrow[t] + 32); }
#pragma unroll
  for (int t = 0; t < 4; ++t) acc[t] = wmma_bf16(a0, b0[t], acc[t]);
#pragma unroll
  for (int t = 0; t < 4; ++t) acc[t] = wmma_bf16(a1, b1[t], acc[t]);
#pragma unroll
  for (int t = 0; t < 4; ++t) {
    int o = (t << 4) + m;
    float bb = b5[o], gg = gamma[o];
#pragma unroll
    for (int r = 0; r < 8; ++r) {
      int row = row0 + r + (ksel << 3);
      int b = row / HsHs, sp = row % HsHs;
      float y = Y[(size_t)row * 64 + o];
      out[((size_t)(b * 64 + o)) * HsHs + sp] = y + (acc[t][r] + bb) * gg;
    }
  }
}

// ---- gate: x1*x2 ------------------------------------------------------------
__global__ void gate_kernel(const bf16* __restrict__ h4, bf16* __restrict__ g, long n) {
  long i = (long)blockIdx.x * blockDim.x + threadIdx.x;
  if (i >= n) return;
  long row = i >> 6; int c = (int)(i & 63);
  g[i] = (bf16)((float)h4[row * 128 + c] * (float)h4[row * 128 + 64 + c]);
}

// ---------------------------------------------------------------------------
extern "C" void kernel_launch(void* const* d_in, const int* in_sizes, int n_in,
                              void* d_out, int out_size, void* d_ws, size_t ws_size,
                              hipStream_t stream) {
  const float* enc0 = (const float*)d_in[0];
  const float* enc1 = (const float*)d_in[1];
  const float* enc2 = (const float*)d_in[2];
  const float* kv_w = (const float*)d_in[3];
  const float* kv_b = (const float*)d_in[4];
  const float* kv_dw_w = (const float*)d_in[5];
  const float* kv_dw_b = (const float*)d_in[6];
  const float* q_w[3]    = {(const float*)d_in[7],  (const float*)d_in[11], (const float*)d_in[15]};
  const float* q_b[3]    = {(const float*)d_in[8],  (const float*)d_in[12], (const float*)d_in[16]};
  const float* q_dw_w[3] = {(const float*)d_in[9],  (const float*)d_in[13], (const float*)d_in[17]};
  const float* q_dw_b[3] = {(const float*)d_in[10], (const float*)d_in[14], (const float*)d_in[18]};
  const float* po_w  = (const float*)d_in[19];
  const float* po_b  = (const float*)d_in[20];
  const float* po1_w = (const float*)d_in[21];
  const float* po1_b = (const float*)d_in[22];
  const float* temp[3] = {(const float*)d_in[23], (const float*)d_in[24], (const float*)d_in[25]};
  const float* c4w[3] = {(const float*)d_in[26], (const float*)d_in[30], (const float*)d_in[34]};
  const float* c4b[3] = {(const float*)d_in[27], (const float*)d_in[31], (const float*)d_in[35]};
  const float* c5w[3] = {(const float*)d_in[28], (const float*)d_in[32], (const float*)d_in[36]};
  const float* c5b[3] = {(const float*)d_in[29], (const float*)d_in[33], (const float*)d_in[37]};
  const float* nqw[3] = {(const float*)d_in[38], (const float*)d_in[40], (const float*)d_in[42]};
  const float* nqb[3] = {(const float*)d_in[39], (const float*)d_in[41], (const float*)d_in[43]};
  const float* n2w[3] = {(const float*)d_in[44], (const float*)d_in[46], (const float*)d_in[48]};
  const float* n2b[3] = {(const float*)d_in[45], (const float*)d_in[47], (const float*)d_in[49]};
  const float* beta[3]  = {(const float*)d_in[50], (const float*)d_in[52], (const float*)d_in[54]};
  const float* gamma[3] = {(const float*)d_in[51], (const float*)d_in[53], (const float*)d_in[55]};
  float* dout = (float*)d_out;

  // ---- workspace carve ----
  char* wsb = (char*)d_ws; size_t off = 0;
  auto alloc = [&](size_t bytes) -> void* {
    void* p = wsb + off; off = (off + bytes + 255) & ~(size_t)255; return p;
  };
  const long MP = (long)Bsz * HWf;                       // 131072 pixels
  bf16* xcat  = (bf16*)alloc((size_t)MP * C3 * 2);       // LN concat
  bf16* xkvp  = (bf16*)alloc((size_t)MP * C6 * 2);       // kv conv1x1 out
  bf16* kkvv  = (bf16*)alloc((size_t)MP * C6 * 2);       // after dwconv (k|v) NHWC
  bf16* kk_cm = (bf16*)alloc((size_t)MP * C3 * 2);       // k-half, channel-major
  bf16* qqp   = (bf16*)alloc((size_t)MP * Cch * 2);      // q conv out (reused)
  bf16* qqn   = (bf16*)alloc((size_t)MP * Cch * 2);      // q dwconv NHWC (reused)
  bf16* qq_cm[3];
  for (int i = 0; i < 3; ++i) qq_cm[i] = (bf16*)alloc((size_t)MP * Cch * 2);
  float* y0 = (float*)alloc((size_t)MP * 64 * 4);
  float* y1 = (float*)alloc((size_t)Bsz * 16384 * 64 * 4);
  float* y2 = (float*)alloc((size_t)Bsz * 4096 * 64 * 4);
  float* ybuf[3] = {y0, y1, y2};
  bf16* lnf = (bf16*)alloc((size_t)MP * 64 * 2);         // reused per branch
  bf16* h4  = (bf16*)alloc((size_t)MP * 128 * 2);
  bf16* gt  = (bf16*)alloc((size_t)MP * 64 * 2);
  float* kpart = (float*)alloc((size_t)Bsz * 256 * C3 * 4);
  float* qpart = (float*)alloc((size_t)Bsz * 256 * 64 * 4);
  float* knorm = (float*)alloc((size_t)Bsz * C3 * 4);
  float* qnorm[3];
  for (int i = 0; i < 3; ++i) qnorm[i] = (float*)alloc((size_t)Bsz * 64 * 4);
  float* gpart = (float*)alloc((size_t)Bsz * 3 * 2 * NCHG * 32 * 96 * 4);
  bf16* mfull = (bf16*)alloc((size_t)Bsz * 3 * 64 * C3 * 2);
  bf16* kvw_bf = (bf16*)alloc((size_t)C6 * C3 * 2);
  bf16* qw_bf[3];  for (int i = 0; i < 3; ++i) qw_bf[i]  = (bf16*)alloc(4096 * 2);
  bf16* c4w_bf[3]; for (int i = 0; i < 3; ++i) c4w_bf[i] = (bf16*)alloc(8192 * 2);
  bf16* c5w_bf[3]; for (int i = 0; i < 3; ++i) c5w_bf[i] = (bf16*)alloc(4096 * 2);
  (void)ws_size; (void)in_sizes; (void)n_in; (void)out_size;

  auto cvt = [&](const float* s, bf16* d, int n) {
    f2bf_kernel<<<(n + 255) / 256, 256, 0, stream>>>(s, d, n);
  };
  cvt(kv_w, kvw_bf, C6 * C3);
  for (int i = 0; i < 3; ++i) { cvt(q_w[i], qw_bf[i], 4096); cvt(c4w[i], c4w_bf[i], 8192); cvt(c5w[i], c5w_bf[i], 4096); }

  // 1) LN + upsample + concat -> xcat [p][192]
  ln_concat_kernel<<<(MP + 255) / 256, 256, 0, stream>>>(
      enc0, enc1, enc2, nqw[0], nqb[0], nqw[1], nqb[1], nqw[2], nqb[2], xcat);

  auto gemm_blocks = [](long M, int N) { long w = (M / 16) * (N / 64); return (int)((w + 7) / 8); };

  // 2) kv = conv1x1(xcat) -> xkvp ; dwconv3 -> kkvv ; transpose k-half
  gemm_bias_bf16<<<gemm_blocks(MP, C6), 256, 0, stream>>>(
      xcat, C3, 0, kvw_bf, C3, kv_b, xkvp, C6, (int)MP, C6, C3);
  {
    long total = MP * C6;
    dwconv3_kernel<<<(int)((total + 255) / 256), 256, 0, stream>>>(xkvp, kv_dw_w, kv_dw_b, kkvv, C6, total);
  }
  {
    dim3 g(HWf / 32, C3 / 32, Bsz), blk(32, 8);
    transpose_bf16<<<g, blk, 0, stream>>>(kkvv, C6, 0, kk_cm, C3);
  }

  // 3) q branches: conv1x1 + dwconv3 + transpose -> qq_cm[br]
  for (int br = 0; br < 3; ++br) {
    gemm_bias_bf16<<<gemm_blocks(MP, 64), 256, 0, stream>>>(
        xcat, C3, 64 * br, qw_bf[br], 64, q_b[br], qqp, 64, (int)MP, 64, 64);
    long total = MP * 64;
    dwconv3_kernel<<<(int)((total + 255) / 256), 256, 0, stream>>>(qqp, q_dw_w[br], q_dw_b[br], qqn, Cch, total);
    dim3 g(HWf / 32, Cch / 32, Bsz), blk(32, 8);
    transpose_bf16<<<g, blk, 0, stream>>>(qqn, Cch, 0, qq_cm[br], Cch);
  }

  // 4) spatial L2 norms of k-channels and q-channels (channel-major streams)
  {
    dim3 g(256, Bsz);
    sqnorm_partial_cm<<<g, C3, 0, stream>>>(kk_cm, kpart, HWf / 256);
    sqnorm_finalize<<<(Bsz * C3 + 255) / 256, 256, 0, stream>>>(kpart, knorm, C3, 256);
    for (int br = 0; br < 3; ++br) {
      sqnorm_partial_cm<<<g, 64, 0, stream>>>(qq_cm[br], qpart, HWf / 256);
      sqnorm_finalize<<<1, Bsz * 64, 0, stream>>>(qpart, qnorm[br], 64, 256);
    }
  }

  // 5) Gram (split-K, contiguous loads) + softmax + Mfull
  {
    dim3 g(NCHG, Bsz, 6);
    gram_kernel<<<g, 384, 0, stream>>>(qq_cm[0], qq_cm[1], qq_cm[2], kk_cm, gpart);
    attn_mfull_kernel<<<Bsz * 6, 96, 0, stream>>>(
        gpart, NCHG, qnorm[0], qnorm[1], qnorm[2], knorm,
        temp[0], temp[1], temp[2], po_w, po1_w, mfull);
  }

  // 6) per-branch: x = Mfull@v + po_b ; y = enc + x*beta ; FFN ; residual out
  const int Hs[3] = {256, 128, 64};
  const int strd[3] = {1, 2, 4};
  const float* encb[3] = {enc0, enc1, enc2};
  const float* pobb[3] = {po_b, po1_b, po1_b};
  const size_t outoff[3] = {0, (size_t)Bsz * 64 * HWf, (size_t)Bsz * 64 * HWf + (size_t)Bsz * 64 * 16384};
  for (int br = 0; br < 3; ++br) {
    int hs = Hs[br], hshs = hs * hs;
    long M = (long)Bsz * hshs;
    {
      long waves = (long)hshs / 16;
      dim3 g((unsigned)((waves + 7) / 8), Bsz);
      attn_out_gemm<<<g, 256, 0, stream>>>(kkvv, mfull, pobb[br], encb[br], beta[br],
                                           ybuf[br], br, strd[br], hs);
    }
    ln_rows_kernel<<<(int)((M + 255) / 256), 256, 0, stream>>>(ybuf[br], n2w[br], n2b[br], lnf, (int)M);
    gemm_bias_bf16<<<gemm_blocks(M, 128), 256, 0, stream>>>(
        lnf, 64, 0, c4w_bf[br], 64, c4b[br], h4, 128, (int)M, 128, 64);
    gate_kernel<<<(int)((M * 64 + 255) / 256), 256, 0, stream>>>(h4, gt, M * 64);
    conv5_out_gemm<<<(int)((M / 16 + 7) / 8), 256, 0, stream>>>(
        gt, c5w_bf[br], c5b[br], ybuf[br], gamma[br], dout + outoff[br], hshs, (int)M);
  }
}
